// DoubleAttention_89051851916054
// MI455X (gfx1250) — matrix operand order repaired and checked by reading, not compiled, not run on hardware
//
#include <hip/hip_runtime.h>
#include <hip/hip_bf16.h>

typedef __bf16 bf16_t;
typedef __attribute__((ext_vector_type(16))) __bf16 v16bf;
typedef __attribute__((ext_vector_type(8)))  float  v8f;

#define HNUM   16
#define HD     64
#define N1C    2048
#define N2C    256
#define NTOT   2304
#define DMODEL 1024

// ---- fragment index helpers (CDNA5 WMMA 16-bit layouts, wave32) ----
// A (16x32, bf16): lane-half g = lane/16, row m = lane%16.
// element e (0..15): K = g*16 + e (contiguous, matches B)
// => pairs p (0..7): kbase = g*16 + p*2  (two consecutive K)
__device__ __forceinline__ int a_pair_kbase(int p, int g) {
    return (g << 4) + (p << 1);
}
// B (32x16, bf16): lane column n = lane%16, element e: K = g*16 + e (contiguous)
// C/D (16x16, f32): vgpr r: row = r + 8*g, column = lane%16

__device__ __forceinline__ v16bf load16bf(const bf16_t* p) {
    union { uint4 u[2]; v16bf v; } cv;
    const uint4* q = (const uint4*)p;
    cv.u[0] = q[0];
    cv.u[1] = q[1];
    return cv.v;
}

// ---------------- weight transpose + fp32 -> bf16 ----------------
__global__ __launch_bounds__(256)
void wtrans_bf16_kernel(const float* __restrict__ W, bf16_t* __restrict__ Wt,
                        int K, int N) {
    int idx = blockIdx.x * blockDim.x + threadIdx.x;
    if (idx >= K * N) return;
    int k = idx / N, n = idx % N;
    Wt[(size_t)n * K + k] = (bf16_t)W[(size_t)k * N + n];
}

// ---------------- activations fp32 -> bf16 (row-major copy) ----------------
__global__ __launch_bounds__(256)
void cvt_bf16_kernel(const float* __restrict__ A, bf16_t* __restrict__ Ab, int n) {
    int idx = blockIdx.x * blockDim.x + threadIdx.x;
    if (idx >= n) return;
    float2 f = *(const float2*)(A + (size_t)idx * 2);
    bf16_t* o = Ab + (size_t)idx * 2;
    o[0] = (bf16_t)f.x; o[1] = (bf16_t)f.y;
}

// ---------------- GEMM: bf16 A (async-LDS staged), bf16 W^T, f32 out (+bias) ------
// One wave computes a 16x64 output block (4 n-tiles), reusing the A fragment 4x.
// A 16x32 tile is staged via double-buffered GLOBAL_LOAD_ASYNC_TO_LDS_B128.
__global__ __launch_bounds__(256)
void gemm_bf16a_kernel(const bf16_t* __restrict__ A, long strideAb,
                       const bf16_t* __restrict__ Wt, const float* __restrict__ bias,
                       float* __restrict__ C, long strideCb,
                       int M, int N, int K) {
    __shared__ bf16_t atile[8][2][16 * 32];   // 8 waves x 2 slots x 1KB
    const int lane = threadIdx.x & 31;
    const int wib = threadIdx.x >> 5;
    const int w = (blockIdx.x * blockDim.x + threadIdx.x) >> 5;
    const int ngrp = N >> 6;                  // 64-wide n groups
    if (w >= (M >> 4) * ngrp) return;         // wave-uniform
    const int mt = w / ngrp, ng = w % ngrp;
    const int g = lane >> 4, c16 = lane & 15;

    const bf16_t* abase = A + (size_t)blockIdx.y * strideAb + (size_t)(mt * 16) * K;
    float* crow = C + (size_t)blockIdx.y * strideCb;

    // per-lane piece of the async tile copy: 2 ops x (8 rows x 4 chunks of 16B)
    const int arow = lane >> 2, achk = lane & 3;
    const unsigned lds0 = (unsigned)(uintptr_t)(void*)&atile[wib][0][0];
    const unsigned lofs = (unsigned)(arow * 64 + achk * 16);
    const bf16_t* g0 = abase + (size_t)arow * K + achk * 8;
    const bf16_t* g1 = g0 + (size_t)8 * K;

    // prologue: stage k-step 0 into slot 0
    {
        unsigned long long a0 = (unsigned long long)(uintptr_t)g0;
        unsigned long long a1 = (unsigned long long)(uintptr_t)g1;
        asm volatile("global_load_async_to_lds_b128 %0, %1, off"
                     :: "v"(lds0 + lofs), "v"(a0) : "memory");
        asm volatile("global_load_async_to_lds_b128 %0, %1, off"
                     :: "v"(lds0 + 512 + lofs), "v"(a1) : "memory");
    }

    const bf16_t* wr0 = Wt + (size_t)(ng * 64 + c16) * K;
    v8f acc[4] = {};
    const int ksteps = K >> 5;
    for (int i = 0; i < ksteps; ++i) {
        const int k0 = i << 5;
        const int slot = i & 1;
        // stage next tile (clamped on the last step) into the other slot
        {
            const int kn = (i + 1 < ksteps) ? (k0 + 32) : k0;
            const unsigned ldsn = lds0 + (unsigned)((slot ^ 1) * 1024);
            unsigned long long a0 = (unsigned long long)(uintptr_t)(g0 + kn);
            unsigned long long a1 = (unsigned long long)(uintptr_t)(g1 + kn);
            asm volatile("global_load_async_to_lds_b128 %0, %1, off"
                         :: "v"(ldsn + lofs), "v"(a0) : "memory");
            asm volatile("global_load_async_to_lds_b128 %0, %1, off"
                         :: "v"(ldsn + 512 + lofs), "v"(a1) : "memory");
        }
        __builtin_prefetch(wr0 + k0 + 256, 0, 0);
        // async loads complete in order: <=2 outstanding => current tile resident
        asm volatile("s_wait_asynccnt 0x2" ::: "memory");
        // A fragment from LDS (row-major 16x32 tile)
        const bf16_t* slotp = &atile[wib][slot][0];
        v16bf af;
        #pragma unroll
        for (int p = 0; p < 8; ++p) {
            const int kbs = a_pair_kbase(p, g);
            unsigned int d = *(const unsigned int*)(slotp + c16 * 32 + kbs);
            union { unsigned int u; bf16_t b[2]; } cv; cv.u = d;
            af[2 * p] = cv.b[0]; af[2 * p + 1] = cv.b[1];
        }
        // 4 B fragments / 4 WMMAs reusing af
        #pragma unroll
        for (int j = 0; j < 4; ++j) {
            v16bf bfr = load16bf(wr0 + (size_t)(j * 16) * K + k0 + g * 16);
            acc[j] = __builtin_amdgcn_wmma_f32_16x16x32_bf16(false, af, false, bfr,
                                                             (short)0, acc[j], false, false);
        }
    }
    asm volatile("s_wait_asynccnt 0x0" ::: "memory");
    #pragma unroll
    for (int j = 0; j < 4; ++j) {
        const float bv = bias[ng * 64 + j * 16 + c16];
        #pragma unroll
        for (int r = 0; r < 8; ++r)
            crow[(size_t)(mt * 16 + r + 8 * g) * N + ng * 64 + j * 16 + c16] = acc[j][r] + bv;
    }
}

// ---------------- LayerNorm + RoPE + pack q/k/vT to bf16 ----------------
__global__ __launch_bounds__(256)
void ln_rope_pack_kernel(const float* __restrict__ qkv1, const float* __restrict__ qkv2,
                         const float* __restrict__ cosb, const float* __restrict__ sinb,
                         const float* __restrict__ qn1w, const float* __restrict__ qn1b,
                         const float* __restrict__ kn1w, const float* __restrict__ kn1b,
                         const float* __restrict__ qn2w, const float* __restrict__ qn2b,
                         const float* __restrict__ kn2w, const float* __restrict__ kn2b,
                         const int* __restrict__ pre_x_ptr,
                         bf16_t* __restrict__ qout, bf16_t* __restrict__ kout,
                         bf16_t* __restrict__ vtout) {
    const int t = blockIdx.x * blockDim.x + threadIdx.x;
    if (t >= 2 * NTOT * HNUM) return;
    const int h = t % HNUM;
    const int n = (t / HNUM) % NTOT;
    const int b = t / (HNUM * NTOT);
    const bool s1 = (n < N1C);
    const float* src = s1 ? (qkv1 + (size_t)(b * N1C + n) * 3 * DMODEL)
                          : (qkv2 + (size_t)(b * N2C + (n - N1C)) * 3 * DMODEL);
    const float* qp = src + h * HD;
    const float* kp = src + DMODEL + h * HD;
    const float* vp = src + 2 * DMODEL + h * HD;

    float qv[HD], kvv[HD];
    float qm = 0.f, km = 0.f;
    #pragma unroll
    for (int d = 0; d < HD; ++d) { qv[d] = qp[d]; kvv[d] = kp[d]; qm += qv[d]; km += kvv[d]; }
    qm *= (1.f / HD); km *= (1.f / HD);
    float qvar = 0.f, kvar = 0.f;
    #pragma unroll
    for (int d = 0; d < HD; ++d) {
        float a = qv[d] - qm; qvar += a * a;
        float c = kvv[d] - km; kvar += c * c;
    }
    const float qr = rsqrtf(qvar * (1.f / HD) + 1e-5f);
    const float kr = rsqrtf(kvar * (1.f / HD) + 1e-5f);
    const float* qw = s1 ? qn1w : qn2w; const float* qb = s1 ? qn1b : qn2b;
    const float* kw = s1 ? kn1w : kn2w; const float* kb = s1 ? kn1b : kn2b;
    #pragma unroll
    for (int d = 0; d < HD; ++d) {
        qv[d]  = (qv[d]  - qm) * qr * qw[d] + qb[d];
        kvv[d] = (kvv[d] - km) * kr * kw[d] + kb[d];
    }
    const int px = *pre_x_ptr;
    if (s1 && n >= px) {
        const int fr = n - px;
        #pragma unroll
        for (int d = 0; d < HD / 2; ++d) {
            const float cc = cosb[fr * (HD / 2) + d];
            const float ss = sinb[fr * (HD / 2) + d];
            float r_ = qv[2 * d], i_ = qv[2 * d + 1];
            qv[2 * d] = r_ * cc - i_ * ss; qv[2 * d + 1] = r_ * ss + i_ * cc;
            r_ = kvv[2 * d]; i_ = kvv[2 * d + 1];
            kvv[2 * d] = r_ * cc - i_ * ss; kvv[2 * d + 1] = r_ * ss + i_ * cc;
        }
    }
    const size_t ob = ((size_t)(b * HNUM + h) * NTOT + n) * HD;
    #pragma unroll
    for (int d = 0; d < HD; ++d) { qout[ob + d] = (bf16_t)qv[d]; kout[ob + d] = (bf16_t)kvv[d]; }
    const size_t vb0 = ((size_t)(b * HNUM + h) * HD) * NTOT + n;
    #pragma unroll
    for (int d = 0; d < HD; ++d) vtout[vb0 + (size_t)d * NTOT] = (bf16_t)vp[d];
}

// ---------------- Flash attention: 1 wave per 16-row q tile ----------------
__global__ __launch_bounds__(256)
void flash_attn_kernel(const bf16_t* __restrict__ qb, const bf16_t* __restrict__ kb,
                       const bf16_t* __restrict__ vtb, bf16_t* __restrict__ ob) {
    __shared__ bf16_t pS[8][16 * 32];       // per-wave P staging (C-layout -> A-layout)
    const int lane = threadIdx.x & 31;
    const int wib = threadIdx.x >> 5;
    const int w = blockIdx.x * 8 + wib;
    const int qtiles = NTOT / 16;
    const int bh = w / qtiles;
    const int qt = w % qtiles;
    const int g = lane >> 4, c16 = lane & 15;
    const float scale = 0.125f;             // hd^-0.5 = 1/8

    const bf16_t* qbase = qb + ((size_t)bh * NTOT + qt * 16) * HD;
    const bf16_t* krow  = kb + (size_t)bh * NTOT * HD;
    const bf16_t* vbase = vtb + (size_t)bh * HD * NTOT;
    bf16_t* psw = pS[wib];

    // Q A-fragments, two 32-wide k halves of hd=64
    v16bf qa[2];
    #pragma unroll
    for (int hb = 0; hb < 2; ++hb) {
        #pragma unroll
        for (int p = 0; p < 8; ++p) {
            const int kbs = a_pair_kbase(p, g);
            unsigned int d = *(const unsigned int*)(qbase + (size_t)c16 * HD + hb * 32 + kbs);
            union { unsigned int u; bf16_t b[2]; } cv; cv.u = d;
            qa[hb][2 * p] = cv.b[0]; qa[hb][2 * p + 1] = cv.b[1];
        }
    }

    float mrow[8], lrow[8];
    v8f accO[4] = {};
    #pragma unroll
    for (int r = 0; r < 8; ++r) { mrow[r] = -1e30f; lrow[r] = 0.f; }

    for (int kv = 0; kv < NTOT; kv += 32) {
        // S = Q K^T for two 16-key tiles (4 WMMAs)
        v8f s[2];
        #pragma unroll
        for (int tkt = 0; tkt < 2; ++tkt) {
            v8f acc = {};
            #pragma unroll
            for (int hb = 0; hb < 2; ++hb) {
                v16bf kf = load16bf(krow + (size_t)(kv + tkt * 16 + c16) * HD + hb * 32 + g * 16);
                acc = __builtin_amdgcn_wmma_f32_16x16x32_bf16(false, qa[hb], false, kf,
                                                              (short)0, acc, false, false);
            }
            s[tkt] = acc;
        }
        // online softmax over the 32 new keys (half-wave reductions match C layout)
        float p0[8], p1[8], corr[8];
        #pragma unroll
        for (int r = 0; r < 8; ++r) {
            const float a = s[0][r] * scale, b = s[1][r] * scale;
            float tm = fmaxf(a, b);
            #pragma unroll
            for (int off = 1; off < 16; off <<= 1) tm = fmaxf(tm, __shfl_xor(tm, off, 32));
            const float mn = fmaxf(mrow[r], tm);
            corr[r] = __expf(mrow[r] - mn);
            mrow[r] = mn;
            p0[r] = __expf(a - mn);
            p1[r] = __expf(b - mn);
            float rs = p0[r] + p1[r];
            #pragma unroll
            for (int off = 1; off < 16; off <<= 1) rs += __shfl_xor(rs, off, 32);
            lrow[r] = lrow[r] * corr[r] + rs;
        }
        #pragma unroll
        for (int j = 0; j < 4; ++j)
            #pragma unroll
            for (int r = 0; r < 8; ++r) accO[j][r] *= corr[r];

        // C-layout P -> LDS (row-major 16x32) -> A-layout fragment
        #pragma unroll
        for (int r = 0; r < 8; ++r) {
            const int row = r + 8 * g;
            psw[row * 32 + c16]      = (bf16_t)p0[r];
            psw[row * 32 + 16 + c16] = (bf16_t)p1[r];
        }
        v16bf pa;
        #pragma unroll
        for (int p = 0; p < 8; ++p) {
            const int kbs = a_pair_kbase(p, g);
            unsigned int d = *(const unsigned int*)(&psw[c16 * 32 + kbs]);
            union { unsigned int u; bf16_t b[2]; } cv; cv.u = d;
            pa[2 * p] = cv.b[0]; pa[2 * p + 1] = cv.b[1];
        }
        // O += P V  (4 WMMAs over hd tiles; V stored transposed [hd][N])
        #pragma unroll
        for (int j = 0; j < 4; ++j) {
            v16bf vf = load16bf(vbase + (size_t)(j * 16 + c16) * NTOT + kv + g * 16);
            accO[j] = __builtin_amdgcn_wmma_f32_16x16x32_bf16(false, pa, false, vf,
                                                              (short)0, accO[j], false, false);
        }
    }

    // finalize: divide by l, scatter heads back into [B][N][D] bf16
    const int b = bh / HNUM, h = bh % HNUM;
    bf16_t* obase = ob + ((size_t)b * NTOT + qt * 16) * DMODEL + h * HD;
    #pragma unroll
    for (int r = 0; r < 8; ++r) {
        const float inv = 1.0f / lrow[r];
        #pragma unroll
        for (int j = 0; j < 4; ++j)
            obase[(size_t)(r + 8 * g) * DMODEL + j * 16 + c16] = (bf16_t)(accO[j][r] * inv);
    }
}

// ---------------- host launch ----------------
extern "C" void kernel_launch(void* const* d_in, const int* in_sizes, int n_in,
                              void* d_out, int out_size, void* d_ws, size_t ws_size,
                              hipStream_t stream) {
    (void)in_sizes; (void)n_in; (void)out_size; (void)ws_size;
    const float* x        = (const float*)d_in[0];
    const float* c        = (const float*)d_in[1];
    const float* fcos     = (const float*)d_in[2];
    const float* fsin     = (const float*)d_in[3];
    const float* W_qkv1   = (const float*)d_in[4];
    const float* b_qkv1   = (const float*)d_in[5];
    const float* W_qkv2   = (const float*)d_in[6];
    const float* b_qkv2   = (const float*)d_in[7];
    const float* qn1w     = (const float*)d_in[8];
    const float* qn1b     = (const float*)d_in[9];
    const float* kn1w     = (const float*)d_in[10];
    const float* kn1b     = (const float*)d_in[11];
    const float* qn2w     = (const float*)d_in[12];
    const float* qn2b     = (const float*)d_in[13];
    const float* kn2w     = (const float*)d_in[14];
    const float* kn2b     = (const float*)d_in[15];
    const float* W_proj1  = (const float*)d_in[16];
    const float* b_proj1  = (const float*)d_in[17];
    const float* W_proj2  = (const float*)d_in[18];
    const float* b_proj2  = (const float*)d_in[19];
    const int*   pre_x    = (const int*)d_in[20];

    char* ws = (char*)d_ws;
    size_t off = 0;
    float*  qkv1 = (float*)(ws + off);  off += (size_t)2 * N1C * 3 * DMODEL * 4;
    float*  qkv2 = (float*)(ws + off);  off += (size_t)2 * N2C * 3 * DMODEL * 4;
    bf16_t* Wt1  = (bf16_t*)(ws + off); off += (size_t)3 * DMODEL * DMODEL * 2;
    bf16_t* Wt2  = (bf16_t*)(ws + off); off += (size_t)3 * DMODEL * DMODEL * 2;
    bf16_t* Wtp1 = (bf16_t*)(ws + off); off += (size_t)DMODEL * DMODEL * 2;
    bf16_t* Wtp2 = (bf16_t*)(ws + off); off += (size_t)DMODEL * DMODEL * 2;
    bf16_t* qbuf = (bf16_t*)(ws + off); off += (size_t)2 * HNUM * NTOT * HD * 2;
    bf16_t* kbuf = (bf16_t*)(ws + off); off += (size_t)2 * HNUM * NTOT * HD * 2;
    bf16_t* vtbf = (bf16_t*)(ws + off); off += (size_t)2 * HNUM * HD * NTOT * 2;
    bf16_t* obuf = (bf16_t*)(ws + off); off += (size_t)2 * NTOT * DMODEL * 2;
    bf16_t* xbf  = (bf16_t*)(ws + off); off += (size_t)2 * N1C * DMODEL * 2;
    bf16_t* cbf  = (bf16_t*)(ws + off); off += (size_t)2 * N2C * DMODEL * 2;
    float* xo = (float*)d_out;
    float* co = (float*)d_out + (size_t)2 * N1C * DMODEL;

    // 1) weights -> bf16 transposed; activations -> bf16
    wtrans_bf16_kernel<<<(DMODEL * 3 * DMODEL) / 256, 256, 0, stream>>>(W_qkv1, Wt1, DMODEL, 3 * DMODEL);
    wtrans_bf16_kernel<<<(DMODEL * 3 * DMODEL) / 256, 256, 0, stream>>>(W_qkv2, Wt2, DMODEL, 3 * DMODEL);
    wtrans_bf16_kernel<<<(DMODEL * DMODEL) / 256, 256, 0, stream>>>(W_proj1, Wtp1, DMODEL, DMODEL);
    wtrans_bf16_kernel<<<(DMODEL * DMODEL) / 256, 256, 0, stream>>>(W_proj2, Wtp2, DMODEL, DMODEL);
    cvt_bf16_kernel<<<(2 * N1C * DMODEL / 2) / 256, 256, 0, stream>>>(x, xbf, 2 * N1C * DMODEL / 2);
    cvt_bf16_kernel<<<(2 * N2C * DMODEL / 2) / 256, 256, 0, stream>>>(c, cbf, 2 * N2C * DMODEL / 2);

    // 2) QKV GEMMs (M=4096 / M=512, N=3072, K=1024), 16x64 per wave
    {
        dim3 gq1(((2 * N1C / 16) * (3 * DMODEL / 64)) / 8, 1);
        gemm_bf16a_kernel<<<gq1, 256, 0, stream>>>(xbf, 0L, Wt1, b_qkv1, qkv1, 0L,
                                                   2 * N1C, 3 * DMODEL, DMODEL);
        dim3 gq2(((2 * N2C / 16) * (3 * DMODEL / 64)) / 8, 1);
        gemm_bf16a_kernel<<<gq2, 256, 0, stream>>>(cbf, 0L, Wt2, b_qkv2, qkv2, 0L,
                                                   2 * N2C, 3 * DMODEL, DMODEL);
    }

    // 3) LN + RoPE + pack
    ln_rope_pack_kernel<<<(2 * NTOT * HNUM) / 256, 256, 0, stream>>>(
        qkv1, qkv2, fcos, fsin, qn1w, qn1b, kn1w, kn1b, qn2w, qn2b, kn2w, kn2b,
        pre_x, qbuf, kbuf, vtbf);

    // 4) flash attention: B*H*(N/16) = 4608 waves -> 576 blocks of 8 waves
    flash_attn_kernel<<<(2 * HNUM * (NTOT / 16)) / 8, 256, 0, stream>>>(qbuf, kbuf, vtbf, obuf);

    // 5) projections (segment-specific weights), grid.y = batch
    {
        dim3 g1(((N1C / 16) * (DMODEL / 64)) / 8, 2);
        gemm_bf16a_kernel<<<g1, 256, 0, stream>>>(obuf, (long)NTOT * DMODEL, Wtp1, b_proj1,
                                                  xo, (long)N1C * DMODEL, N1C, DMODEL, DMODEL);
        dim3 g2(((N2C / 16) * (DMODEL / 64)) / 8, 2);
        gemm_bf16a_kernel<<<g2, 256, 0, stream>>>(obuf + (size_t)N1C * DMODEL, (long)NTOT * DMODEL,
                                                  Wtp2, b_proj2,
                                                  co, (long)N2C * DMODEL, N2C, DMODEL, DMODEL);
    }
}
